// AttnMPNN_38517266710685
// MI455X (gfx1250) — matrix operand order repaired and checked by reading, not compiled, not run on hardware
//
#include <hip/hip_runtime.h>

#define N_NODES 50000
#define N_EDGES 800000
#define D 64
#define HID 64

typedef __attribute__((ext_vector_type(16))) _Float16 v16h;
typedef __attribute__((ext_vector_type(8)))  float    v8f;

// order-preserving float <-> uint encoding for atomicMax over signed floats
__device__ __forceinline__ unsigned enc_f32(float f) {
  unsigned u = __float_as_uint(f);
  return (u & 0x80000000u) ? ~u : (u | 0x80000000u);
}
__device__ __forceinline__ float dec_f32(unsigned e) {
  unsigned u = (e & 0x80000000u) ? (e ^ 0x80000000u) : ~e;
  return __uint_as_float(u);
}

// ---------------- k0: zero segment state ----------------
__global__ void k0_init(float* zbase, int n) {
  int i = blockIdx.x * blockDim.x + threadIdx.x;
  if (i < n) zbase[i] = 0.0f;
}

// ---------------- k1: per-edge attention logits via WMMA ----------------
// one wave handles 16 edges; A = [nf[src] | nf[dst]] (16x128), B = W_attn (128x64)
__global__ void __launch_bounds__(256) k1_edge_logits(
    const float* __restrict__ nf, const int* __restrict__ src, const int* __restrict__ dst,
    const float* __restrict__ W_attn, const float* __restrict__ b_attn,
    const float* __restrict__ w_fc,
    float* __restrict__ logits, unsigned* __restrict__ segmax)
{
  // stage W_attn transposed in LDS as fp16: Wt[col][k], 64*128 halves = 16 KB
  __shared__ __align__(32) _Float16 Wt[HID * 2 * D];
  for (int i = threadIdx.x; i < HID * 2 * D; i += 256) {
    int c = i >> 7, k = i & 127;
    Wt[i] = (_Float16)W_attn[k * HID + c];
  }
  __syncthreads();

  const int lane = threadIdx.x & 31;
  const int wv   = threadIdx.x >> 5;
  const int m    = lane & 15;   // A row / B column / C column
  const int hi   = lane >> 4;   // lane half selects K sub-range
  const long ebase = (long)(blockIdx.x * 8 + wv) * 16;  // grid sized so ebase < E always

  const int e_row = (int)ebase + m;
  const int sN = src[e_row];
  const int dN = dst[e_row];

  // A tiles: 4 chunks of K=32. A k order chosen here: per lane 16 contiguous
  // K elements at Klocal = hi*16 (matches B).
  v16h A[4];
#pragma unroll
  for (int kc = 0; kc < 4; ++kc) {
    const float* rowp = (kc < 2) ? (nf + (long)sN * D) : (nf + (long)dN * D);
    const int ko = (kc & 1) * 32 + hi * 16;
#pragma unroll
    for (int j = 0; j < 8; ++j) A[kc][j]     = (_Float16)rowp[ko + j];
#pragma unroll
    for (int j = 0; j < 8; ++j) A[kc][8 + j] = (_Float16)rowp[ko + 8 + j];
  }

  float partial[8];
#pragma unroll
  for (int r = 0; r < 8; ++r) partial[r] = 0.0f;

#pragma unroll
  for (int t = 0; t < 4; ++t) {           // HID column tiles
    const float bcol = b_attn[t * 16 + m];
    v8f acc;
#pragma unroll
    for (int r = 0; r < 8; ++r) acc[r] = bcol;   // fuse bias into C
#pragma unroll
    for (int kc = 0; kc < 4; ++kc) {
      // B 32x16 fp16 as loaded here: lane = column, 16 contiguous K halves at hi*16 (same k order as A)
      const v16h B = *(const v16h*)&Wt[(t * 16 + m) * 128 + kc * 32 + hi * 16];
      acc = __builtin_amdgcn_wmma_f32_16x16x32_f16(
          false, A[kc], false, B, (short)0, acc, false, false);
    }
    const float wc = w_fc[t * 16 + m];
#pragma unroll
    for (int r = 0; r < 8; ++r) {
      float h = acc[r] > 0.0f ? acc[r] : 0.0f;   // ReLU
      partial[r] += h * wc;                      // fold w_fc dot
    }
  }

  // butterfly-reduce partials across the 16 lanes of each half
#pragma unroll
  for (int off = 1; off < 16; off <<= 1) {
#pragma unroll
    for (int r = 0; r < 8; ++r) partial[r] += __shfl_xor(partial[r], off, 32);
  }

  if (m < 8) {  // 16 writer lanes: rows 0-7 from half 0, rows 8-15 from half 1
    const int  row = hi * 8 + m;
    const long e   = ebase + row;
    float l = partial[m];
    l = (l >= 0.0f) ? l : 0.01f * l;             // LeakyReLU
    logits[e] = l;
    atomicMax(&segmax[dst[e]], enc_f32(l));
  }
}

// ---------------- k2a: exp + segment sum / degree ----------------
__global__ void k2a_exp(const float* __restrict__ logits, const int* __restrict__ dst,
                        const unsigned* __restrict__ segmax,
                        float* __restrict__ expv, float* __restrict__ segsum,
                        float* __restrict__ deg)
{
  int e = blockIdx.x * blockDim.x + threadIdx.x;
  if (e < N_EDGES) {
    int d = dst[e];
    float ev = __expf(logits[e] - dec_f32(segmax[d]));
    expv[e] = ev;
    atomicAdd(&segsum[d], ev);
    atomicAdd(&deg[d], 1.0f);
  }
}

// ---------------- k2b: scatter nf[src]*e into agg ----------------
__global__ void k2b_scatter(const float* __restrict__ nf, const int* __restrict__ src,
                            const int* __restrict__ dst, const float* __restrict__ expv,
                            float* __restrict__ agg)
{
  long idx = (long)blockIdx.x * blockDim.x + threadIdx.x;
  if (idx < (long)N_EDGES * 4) {
    int e = (int)(idx >> 2);
    int q = (int)(idx & 3);
    float ev = expv[e];
    const float* rp = nf  + (long)src[e] * D + q * 16;
    float*       ap = agg + (long)dst[e] * D + q * 16;
#pragma unroll
    for (int j = 0; j < 16; ++j) atomicAdd(&ap[j], rp[j] * ev);
  }
}

// ---------------- k3: node GEMM via WMMA ----------------
__global__ void __launch_bounds__(256) k3_node(
    const float* __restrict__ nf, const float* __restrict__ agg,
    const float* __restrict__ segsum, const float* __restrict__ deg,
    const float* __restrict__ W_node, const float* __restrict__ b_node,
    float* __restrict__ out)
{
  __shared__ __align__(32) _Float16 Wt[D * 2 * D];
  for (int i = threadIdx.x; i < D * 2 * D; i += 256) {
    int c = i >> 7, k = i & 127;
    Wt[i] = (_Float16)W_node[k * D + c];
  }
  __syncthreads();

  const int lane = threadIdx.x & 31;
  const int wv   = threadIdx.x >> 5;
  const int m    = lane & 15;
  const int hi   = lane >> 4;
  const long nbase = (long)(blockIdx.x * 8 + wv) * 16;
  if (nbase >= N_NODES) return;   // wave-uniform guard (3125 tiles total)

  const int v = (int)nbase + m;
  const float dg = deg[v];
  // attn = e/segsum, mean divides by max(deg,1); deg==0 -> agg_m = 0
  const float scale = (dg > 0.0f) ? 1.0f / (segsum[v] * dg) : 0.0f;

  v16h A[4];
#pragma unroll
  for (int kc = 0; kc < 4; ++kc) {
    const bool isAgg = (kc >= 2);
    const float* rowp = isAgg ? (agg + (long)v * D) : (nf + (long)v * D);
    const float mul = isAgg ? scale : 1.0f;
    const int ko = (kc & 1) * 32 + hi * 16;
#pragma unroll
    for (int j = 0; j < 8; ++j) A[kc][j]     = (_Float16)(rowp[ko + j] * mul);
#pragma unroll
    for (int j = 0; j < 8; ++j) A[kc][8 + j] = (_Float16)(rowp[ko + 8 + j] * mul);
  }

#pragma unroll
  for (int t = 0; t < 4; ++t) {
    const float bcol = b_node[t * 16 + m];
    v8f acc;
#pragma unroll
    for (int r = 0; r < 8; ++r) acc[r] = bcol;
#pragma unroll
    for (int kc = 0; kc < 4; ++kc) {
      const v16h B = *(const v16h*)&Wt[(t * 16 + m) * 128 + kc * 32 + hi * 16];
      acc = __builtin_amdgcn_wmma_f32_16x16x32_f16(
          false, A[kc], false, B, (short)0, acc, false, false);
    }
    // D layout: VGPR r holds row (r + 8*hi), column = m, col tile t
#pragma unroll
    for (int r = 0; r < 8; ++r)
      out[(nbase + r + hi * 8) * D + t * 16 + m] = acc[r];
  }
}

// ---------------- launch ----------------
extern "C" void kernel_launch(void* const* d_in, const int* in_sizes, int n_in,
                              void* d_out, int out_size, void* d_ws, size_t ws_size,
                              hipStream_t stream) {
  (void)in_sizes; (void)n_in; (void)out_size; (void)ws_size;
  const float* nf     = (const float*)d_in[0];
  const int*   src    = (const int*)  d_in[1];
  const int*   dst    = (const int*)  d_in[2];
  const float* W_attn = (const float*)d_in[3];
  const float* b_attn = (const float*)d_in[4];
  const float* w_fc   = (const float*)d_in[5];
  const float* W_node = (const float*)d_in[6];
  const float* b_node = (const float*)d_in[7];
  float* out = (float*)d_out;

  // workspace layout
  char* ws = (char*)d_ws;
  float*    logits = (float*)ws;                                   // E floats
  float*    expv   = (float*)(ws + (size_t)N_EDGES * 4);           // E floats
  unsigned* segmax = (unsigned*)(ws + (size_t)N_EDGES * 8);        // N uints
  float*    segsum = (float*)(ws + (size_t)N_EDGES * 8 + (size_t)N_NODES * 4);
  float*    deg    = (float*)(ws + (size_t)N_EDGES * 8 + (size_t)N_NODES * 8);
  float*    agg    = (float*)(ws + (size_t)N_EDGES * 8 + (size_t)N_NODES * 12);

  // zero segmax(0 encodes -NaN, below every finite logit) + segsum + deg + agg
  const int nzero = N_NODES * (3 + D);
  k0_init<<<(nzero + 255) / 256, 256, 0, stream>>>((float*)segmax, nzero);

  // edge logits: E/16 = 50000 waves, 8 waves/block -> 6250 blocks
  k1_edge_logits<<<N_EDGES / 16 / 8, 256, 0, stream>>>(
      nf, src, dst, W_attn, b_attn, w_fc, logits, segmax);

  k2a_exp<<<(N_EDGES + 255) / 256, 256, 0, stream>>>(
      logits, dst, segmax, expv, segsum, deg);

  k2b_scatter<<<(N_EDGES * 4 + 255) / 256, 256, 0, stream>>>(
      nf, src, dst, expv, agg);

  // node GEMM: N/16 = 3125 waves -> 391 blocks (last block partially active)
  k3_node<<<(N_NODES / 16 + 7) / 8, 256, 0, stream>>>(
      nf, agg, segsum, deg, W_node, b_node, out);
}
